// VNet_Decouple_Attention_ABC_31834297598522
// MI455X (gfx1250) — matrix operand order repaired and checked by reading, not compiled, not run on hardware
//
#include <hip/hip_runtime.h>
#include <hip/hip_bf16.h>
#include <math.h>

// ---------------------------------------------------------------------------
// VNet decoupled cross-attention for MI455X (gfx1250, wave32, WMMA).
// B2=8 (two halves of 4), N=256, C=256, CH=2048, M=B*CH=8192.
//   k0: Q/K/V projections  (bf16 WMMA GEMM, f32->bf16 on load, b128 A loads)
//   k1: per-(dir,b) sum/sumsq of scores (WMMA recompute, 16 tiles/wave)
//   k2: flash pass: kf tile staged to LDS by all 4 waves (coalesced b128),
//       wave0 computes scores + globally-normalized online softmax,
//       P tile transposed through LDS, all waves run PV WMMAs
//   k3: out = ctx @ Wo (f32 out)
// Working set (~32 MB ws) is fully L2-resident; no 268 MB scores tensor.
// ---------------------------------------------------------------------------

typedef __bf16 bf16_t;
typedef __attribute__((ext_vector_type(16))) __bf16 v16bf;
typedef __attribute__((ext_vector_type(8)))  __bf16 v8bf;
typedef __attribute__((ext_vector_type(2)))  __bf16 v2bf;
typedef __attribute__((ext_vector_type(8)))  float  v8f;

#define NTOK 256
#define CDIM 256
#define CHD  2048
#define MDIM 8192
#define EPSV 1e-5f

// ---- WMMA fragment loaders (CDNA5 16x16x32 bf16 layouts) -------------------
// A (16x32): lane r holds row r; half h: K in [h*16, h*16+16).
// Generic strided gather (for transposed views):
__device__ __forceinline__ v16bf a_frag_bf16(const bf16_t* base, int rs, int cs) {
  int lane = threadIdx.x & 31, r = lane & 15, h = lane >> 4;
  v16bf a;
#pragma unroll
  for (int j = 0; j < 8; ++j) {
    int k = h*16 + 2*j;
    a[2*j]   = base[r*rs + k*cs];
    a[2*j+1] = base[r*rs + (k+1)*cs];
  }
  return a;
}
// Fast path, cs==1: two contiguous 16B chunks per lane.
__device__ __forceinline__ v16bf a_frag_bf16_row(const bf16_t* base, int rs) {
  int lane = threadIdx.x & 31, r = lane & 15, h = lane >> 4;
  const bf16_t* p = base + (size_t)r * rs + h * 16;
  v8bf lo = *(const v8bf*)(p);
  v8bf hi = *(const v8bf*)(p + 8);
  return __builtin_shufflevector(lo, hi, 0,1,2,3,4,5,6,7,8,9,10,11,12,13,14,15);
}
// Fast path, f32 source, cs==1: two 32B chunks + pack-convert.
__device__ __forceinline__ v16bf a_frag_f32_row(const float* base, int rs) {
  int lane = threadIdx.x & 31, r = lane & 15, h = lane >> 4;
  const float* p = base + (size_t)r * rs + h * 16;
  v8f lo = *(const v8f*)(p);
  v8f hi = *(const v8f*)(p + 8);
  v16bf a;
#pragma unroll
  for (int j = 0; j < 8; ++j) { a[j] = (bf16_t)lo[j]; a[8+j] = (bf16_t)hi[j]; }
  return a;
}
// B (32x16): lane r holds column r; K = h*16 + idx.
__device__ __forceinline__ v16bf b_frag_bf16(const bf16_t* base, int rs, int cs) {
  int lane = threadIdx.x & 31, r = lane & 15, h = lane >> 4;
  v16bf b;
#pragma unroll
  for (int j = 0; j < 8; ++j) {
    int k = h*16 + 2*j;
    b[2*j]   = base[k*rs + r*cs];
    b[2*j+1] = base[(k+1)*rs + r*cs];
  }
  return b;
}
// Fast path, rs==1: 16 contiguous K values (32B) per lane.
__device__ __forceinline__ v16bf b_frag_bf16_kcontig(const bf16_t* base, int cs) {
  int lane = threadIdx.x & 31, r = lane & 15, h = lane >> 4;
  const bf16_t* p = base + (size_t)r * cs + h * 16;
  v8bf lo = *(const v8bf*)(p);
  v8bf hi = *(const v8bf*)(p + 8);
  return __builtin_shufflevector(lo, hi, 0,1,2,3,4,5,6,7,8,9,10,11,12,13,14,15);
}
__device__ __forceinline__ v16bf b_frag_f32(const float* base, int rs, int cs) {
  int lane = threadIdx.x & 31, r = lane & 15, h = lane >> 4;
  v16bf b;
#pragma unroll
  for (int j = 0; j < 8; ++j) {
    int k = h*16 + 2*j;
    b[2*j]   = (bf16_t)base[k*rs + r*cs];
    b[2*j+1] = (bf16_t)base[(k+1)*rs + r*cs];
  }
  return b;
}
__device__ __forceinline__ v8f wmma_bf16(v16bf a, v16bf b, v8f c) {
  return __builtin_amdgcn_wmma_f32_16x16x32_bf16(false, a, false, b, (short)0, c,
                                                 false, false);
}

// ---- k0: C(bf16, MxN) = A(f32, MxK) @ B(f32, KxN) --------------------------
__global__ __launch_bounds__(128) void gemm_f32f32_bf16(
    const float* __restrict__ A, const float* __restrict__ B,
    bf16_t* __restrict__ C, int M, int N, int K) {
  int wave = threadIdx.x >> 5;
  int m0 = blockIdx.x * 16;
  int n0 = (blockIdx.y * 4 + wave) * 16;
  if (m0 >= M || n0 >= N) return;
  v8f acc = {};
  for (int k0 = 0; k0 < K; k0 += 32) {
    v16bf a = a_frag_f32_row(A + (size_t)m0*K + k0, K);
    v16bf b = b_frag_f32(B + (size_t)k0*N + n0, N, 1);
    acc = wmma_bf16(a, b, acc);
  }
  int lane = threadIdx.x & 31, r = lane & 15, h = lane >> 4;
#pragma unroll
  for (int i = 0; i < 8; ++i)
    C[(size_t)(m0 + i + 8*h)*N + n0 + r] = (bf16_t)acc[i];
}

// ---- k3: C(f32, MxN) = A(bf16, MxK) @ B(f32, KxN) --------------------------
__global__ __launch_bounds__(128) void gemm_bf16f32_f32(
    const bf16_t* __restrict__ A, const float* __restrict__ B,
    float* __restrict__ C, int M, int N, int K) {
  int wave = threadIdx.x >> 5;
  int m0 = blockIdx.x * 16;
  int n0 = (blockIdx.y * 4 + wave) * 16;
  if (m0 >= M || n0 >= N) return;
  v8f acc = {};
  for (int k0 = 0; k0 < K; k0 += 32) {
    v16bf a = a_frag_bf16_row(A + (size_t)m0*K + k0, K);
    v16bf b = b_frag_f32(B + (size_t)k0*N + n0, N, 1);
    acc = wmma_bf16(a, b, acc);
  }
  int lane = threadIdx.x & 31, r = lane & 15, h = lane >> 4;
#pragma unroll
  for (int i = 0; i < 8; ++i)
    C[(size_t)(m0 + i + 8*h)*N + n0 + r] = acc[i];
}

// ---- k1: per-(dir,b) sum & sumsq of scores; 16 m-tiles per wave ------------
// scores[c,m] = sum_n Q[qb,n,c] * K[kvb + m/CH, n, m%CH]
__global__ __launch_bounds__(32) void stats_kernel(
    const bf16_t* __restrict__ Qbf, const bf16_t* __restrict__ Kbf,
    float* __restrict__ stats) {
  int z = blockIdx.z;                 // z = d*4 + b
  int d = z >> 2, b = z & 3;
  int qb  = (d == 0) ? b : 4 + b;     // d=0: l2u (q from lower half)
  int kvb = (d == 0) ? 4 : 0;
  int c0 = blockIdx.y * 16;
  int mbase = blockIdx.x * 256;       // 16 tiles of 16 along m
  const bf16_t* qp = Qbf + (size_t)qb * NTOK * CHD;

  v16bf aq[8];                        // q^T A-frags: invariant over m
#pragma unroll
  for (int kk = 0; kk < 8; ++kk)
    aq[kk] = a_frag_bf16(qp + (size_t)(kk*32)*CHD + c0, 1, CHD);

  float s1 = 0.f, s2 = 0.f;
  for (int mt = 0; mt < 16; ++mt) {
    int m0 = mbase + mt * 16;
    int mb = m0 / CHD, cm = m0 % CHD; // 16-tiles never straddle CH
    const bf16_t* kp = Kbf + (size_t)(kvb + mb) * NTOK * CHD;
    v8f acc = {};
#pragma unroll
    for (int kk = 0; kk < 8; ++kk) {
      v16bf bk = b_frag_bf16(kp + (size_t)(kk*32)*CHD + cm, CHD, 1);
      acc = wmma_bf16(aq[kk], bk, acc);
    }
#pragma unroll
    for (int i = 0; i < 8; ++i) { s1 += acc[i]; s2 += acc[i]*acc[i]; }
  }
#pragma unroll
  for (int off = 16; off >= 1; off >>= 1) {
    s1 += __shfl_xor(s1, off, 32);
    s2 += __shfl_xor(s2, off, 32);
  }
  if ((threadIdx.x & 31) == 0) {
    atomicAdd(&stats[2*z],     s1);
    atomicAdd(&stats[2*z + 1], s2);
  }
}

// ---- k2: flash pass. 4 waves per (d,b,c-tile). ------------------------------
__global__ __launch_bounds__(128) void flash_kernel(
    const bf16_t* __restrict__ Qbf, const bf16_t* __restrict__ Kbf,
    const bf16_t* __restrict__ Vbf, const float* __restrict__ stats,
    bf16_t* __restrict__ ctx) {
  __shared__ __align__(16) bf16_t kLDS[NTOK * 32];   // 16 KB staged kf tile
  __shared__ __align__(16) bf16_t pTile[16 * 32];    // P transpose bounce
  __shared__ float scLDS[16];
  __shared__ float sumLDS[16];

  int z = blockIdx.y, d = z >> 2, b = z & 3;
  int qb  = (d == 0) ? b : 4 + b;
  int kvb = (d == 0) ? 4 : 0;
  int c0 = blockIdx.x * 16;

  const float CNT = (float)CHD * (float)MDIM;
  float mu   = stats[2*z] / CNT;
  float var  = stats[2*z + 1] / CNT - mu * mu;
  float rstd = rsqrtf(var + EPSV);
  float beta = -mu * rstd;

  int tid = threadIdx.x, wave = tid >> 5, lane = tid & 31;
  int r = lane & 15, h = lane >> 4;

  const bf16_t* qp = Qbf + (size_t)qb * NTOK * CHD;

  v8f acc[4] = {};                       // ctx tiles, n-tiles t = wave*4+tl
  float rowMax[8], rowSum[8];
#pragma unroll
  for (int i = 0; i < 8; ++i) { rowMax[i] = -__builtin_inff(); rowSum[i] = 0.f; }

  v16bf aq[8];                           // q^T A-frags (wave 0 only)
  if (wave == 0) {
#pragma unroll
    for (int kk = 0; kk < 8; ++kk)
      aq[kk] = a_frag_bf16(qp + (size_t)(kk*32)*CHD + c0, 1, CHD);
  }

  for (int m0 = 0; m0 < MDIM; m0 += 32) {
    int mb = m0 / CHD, cm = m0 % CHD;    // 32 | CHD -> no straddle
    const bf16_t* kp = Kbf + (size_t)(kvb + mb) * NTOK * CHD;
    const bf16_t* vp = Vbf + (size_t)(kvb + mb) * NTOK * CHD;

    // Cooperative stage of kf tile (256 n x 32 m) with coalesced 16B loads.
#pragma unroll
    for (int it = 0; it < 8; ++it) {
      int chunk = tid + (it << 7);       // 0..1023, 16B each
      int row = chunk >> 2, part = chunk & 3;
      *(v8bf*)(kLDS + row*32 + part*8) =
          *(const v8bf*)(kp + (size_t)row*CHD + cm + part*8);
    }
    // Prefetch next m-tile of kf while wave0 crunches (global_prefetch_b8).
    if (wave == 3 && m0 + 32 < MDIM) {
      int m1 = m0 + 32, mb1 = m1 / CHD, cm1 = m1 % CHD;
      const bf16_t* kpn = Kbf + (size_t)(kvb + mb1) * NTOK * CHD;
      __builtin_prefetch(kpn + (size_t)lane * 8 * CHD + cm1, 0, 1);
    }
    __syncthreads();

    if (wave == 0) {
      v8f s0 = {}, s1 = {};
#pragma unroll
      for (int kk = 0; kk < 8; ++kk) {
        const bf16_t* kbase = kLDS + (kk*32)*32;
        v16bf b0 = b_frag_bf16(kbase,      32, 1);
        v16bf b1 = b_frag_bf16(kbase + 16, 32, 1);
        s0 = wmma_bf16(aq[kk], b0, s0);
        s1 = wmma_bf16(aq[kk], b1, s1);
      }
#pragma unroll
      for (int i = 0; i < 8; ++i) {
        float x0 = s0[i]*rstd + beta;    // globally-normalized score
        float x1 = s1[i]*rstd + beta;
        float mx = fmaxf(x0, x1);
#pragma unroll
        for (int msk = 8; msk >= 1; msk >>= 1)   // row max (half-wave)
          mx = fmaxf(mx, __shfl_xor(mx, msk, 16));
        float nm = fmaxf(rowMax[i], mx);
        float sc = __expf(rowMax[i] - nm);
        rowMax[i] = nm;
        float p0 = __expf(x0 - nm);
        float p1 = __expf(x1 - nm);
        float ps = p0 + p1;
#pragma unroll
        for (int msk = 8; msk >= 1; msk >>= 1)   // row sum (half-wave)
          ps += __shfl_xor(ps, msk, 16);
        rowSum[i] = rowSum[i]*sc + ps;
        int row = i + 8*h;               // D-layout row -> A-layout source
        pTile[row*32 + r]      = (bf16_t)p0;
        pTile[row*32 + r + 16] = (bf16_t)p1;
        if (r == 0) scLDS[row] = sc;
      }
    }
    __syncthreads();

    v16bf ap = a_frag_bf16_row(pTile, 32);       // ds_load b128 pairs
#pragma unroll
    for (int tl = 0; tl < 4; ++tl) {
#pragma unroll
      for (int i = 0; i < 8; ++i)
        acc[tl][i] *= scLDS[i + 8*h];            // flash rescale
    }
#pragma unroll
    for (int tl = 0; tl < 4; ++tl) {
      int t = wave*4 + tl;
      v16bf bv = b_frag_bf16_kcontig(vp + (size_t)(t*16)*CHD + cm, CHD); // vf^T
      acc[tl] = wmma_bf16(ap, bv, acc[tl]);
    }
    __syncthreads();
  }

  if (wave == 0 && r == 0) {
#pragma unroll
    for (int i = 0; i < 8; ++i) sumLDS[i + 8*h] = rowSum[i];
  }
  __syncthreads();

#pragma unroll
  for (int tl = 0; tl < 4; ++tl) {
    int t = wave*4 + tl;
    int ncol = t*16 + r;
#pragma unroll
    for (int i = 0; i < 8; i += 2) {             // paired 4B ctx stores
      int crow = c0 + i + 8*h;
      v2bf pv;
      pv[0] = (bf16_t)(acc[tl][i]     / sumLDS[i     + 8*h]);
      pv[1] = (bf16_t)(acc[tl][i + 1] / sumLDS[i + 1 + 8*h]);
      *(v2bf*)(&ctx[((size_t)z * NTOK + ncol) * CHD + crow]) = pv;
    }
  }
}

extern "C" void kernel_launch(void* const* d_in, const int* in_sizes, int n_in,
                              void* d_out, int out_size, void* d_ws, size_t ws_size,
                              hipStream_t stream) {
  (void)in_sizes; (void)n_in; (void)out_size; (void)ws_size;
  const float* emb = (const float*)d_in[0];
  const float* Wq  = (const float*)d_in[1];
  const float* Wk  = (const float*)d_in[2];
  const float* Wv  = (const float*)d_in[3];
  const float* Wo  = (const float*)d_in[4];
  float* out = (float*)d_out;

  // Workspace: Q/K/V bf16 (8 MB each) + ctx bf16 (8 MB) + 16 stat floats.
  size_t qkvElems = (size_t)8 * NTOK * CHD;            // 4M elems each
  bf16_t* Qbf = (bf16_t*)d_ws;
  bf16_t* Kbf = Qbf + qkvElems;
  bf16_t* Vbf = Kbf + qkvElems;
  bf16_t* ctx = Vbf + qkvElems;                        // 2*4*256*2048 elems
  float* stats = (float*)(ctx + (size_t)2 * 4 * NTOK * CHD);

  hipMemsetAsync(stats, 0, 16 * sizeof(float), stream);

  // k0: projections — emb viewed as (2048 x 256) @ W (256 x 2048)
  dim3 blk(128);
  dim3 gproj(2048/16, 2048/64);
  gemm_f32f32_bf16<<<gproj, blk, 0, stream>>>(emb, Wq, Qbf, 2048, CHD, CDIM);
  gemm_f32f32_bf16<<<gproj, blk, 0, stream>>>(emb, Wk, Kbf, 2048, CHD, CDIM);
  gemm_f32f32_bf16<<<gproj, blk, 0, stream>>>(emb, Wv, Vbf, 2048, CHD, CDIM);

  // k1: score sum/sumsq per (direction, b); 16 m-tiles per wave
  dim3 gstats(MDIM/256, CHD/16, 8);
  stats_kernel<<<gstats, dim3(32), 0, stream>>>(Qbf, Kbf, stats);

  // k2: normalized softmax + ctx accumulation (flash style)
  dim3 gflash(CHD/16, 8);
  flash_kernel<<<gflash, dim3(128), 0, stream>>>(Qbf, Kbf, Vbf, stats, ctx);

  // k3: out = ctx (2048 x 2048) @ Wo (2048 x 256); row order == concat output.
  dim3 gfin(2048/16, 256/64);
  gemm_bf16f32_f32<<<gfin, blk, 0, stream>>>(ctx, Wo, out, 2048, 256, CHD);
}
